// InterestEvolutionLayer_31396210933838
// MI455X (gfx1250) — compile-verified, matrix operand order checked by reading, not run on hardware
//
#include <hip/hip_runtime.h>
#include <hip/hip_bf16.h>

#define Bb 4096
#define Tt 200
#define Ee 64

typedef __attribute__((ext_vector_type(16))) __bf16 v16bf;
typedef __attribute__((ext_vector_type(8)))  float  v8f;

#define WMMA_BF16(Aop, Bop, Cacc) \
  __builtin_amdgcn_wmma_f32_16x16x32_bf16(false, (Aop), false, (Bop), (short)0, (Cacc), false, false)

__device__ __forceinline__ float sigf(float x) {
  return 1.0f / (1.0f + __expf(-x));
}
__device__ __forceinline__ float tanhfast(float x) {
  float a = fabsf(x);
  float e = __expf(-2.0f * a);
  float r = (1.0f - e) / (1.0f + e);
  return copysignf(r, x);
}

// --- gfx1250 async global->LDS copy (ASYNCcnt-tracked), 16 bytes per lane ---
__device__ __forceinline__ void async_copy16(void* lds, const void* g) {
  unsigned ldsoff = (unsigned)(unsigned long long)(size_t)lds;  // addr[31:0] = LDS offset
  asm volatile("global_load_async_to_lds_b128 %0, %1, off"
               :
               : "v"(ldsoff), "v"(g)
               : "memory");
}
__device__ __forceinline__ void wait_async0() {
  asm volatile("s_wait_asynccnt 0x0" ::: "memory");
}
__device__ __forceinline__ void wait_async2() {
  asm volatile("s_wait_asynccnt 0x2" ::: "memory");
}

// ---------------------------------------------------------------------------
// Kernel 0: swizzle aw1 (256x32), W (64x192), U (64x192) fp32 -> bf16 in the
// WMMA B-operand layout: operand (ntile, kchunk) -> 32 lanes x 16 bf16 slots.
//   n = ntile*16 + (lane&15)
//   k = kchunk*32 + 8*(lane>=16) + slot + (slot>=8 ? 8 : 0)
// ---------------------------------------------------------------------------
__global__ __launch_bounds__(256) void prep_swizzle(
    const float* __restrict__ aw1, const float* __restrict__ W,
    const float* __restrict__ U, __bf16* __restrict__ oA1,
    __bf16* __restrict__ oW, __bf16* __restrict__ oU) {
  int idx = blockIdx.x * 256 + threadIdx.x;  // total 40960
  if (idx < 16384) {  // aw1: nt(2) x kc(8) x lane(32) x slot(16)
    int s = idx & 15, L = (idx >> 4) & 31, kc = (idx >> 9) & 7, nt = idx >> 12;
    int k = kc * 32 + ((L >= 16) ? 8 : 0) + s + ((s >= 8) ? 8 : 0);
    int n = nt * 16 + (L & 15);
    oA1[idx] = (__bf16)aw1[k * 32 + n];
  } else {  // W/U: nt(12) x kc(2) x lane(32) x slot(16) = 12288 each
    int which = (idx < 28672) ? 0 : 1;
    int i2 = idx - (which ? 28672 : 16384);
    int s = i2 & 15, L = (i2 >> 4) & 31, kc = (i2 >> 9) & 1, nt = i2 >> 10;
    int k = kc * 32 + ((L >= 16) ? 8 : 0) + s + ((s >= 8) ? 8 : 0);
    int n = nt * 16 + (L & 15);
    const float* src = which ? U : W;
    (which ? oU : oW)[i2] = (__bf16)src[k * 192 + n];
  }
}

// ---------------------------------------------------------------------------
// Kernel 1: attention logits. One wave per 16-row M-tile of flattened (b,t).
// feat(16x256) @ aw1(256x32) via 16x v_wmma_f32_16x16x32_bf16, then VALU tail
// MLP 32 -> 16 -> 1 through an LDS transpose.
// ---------------------------------------------------------------------------
__global__ __launch_bounds__(256) void attn_logits(
    const float* __restrict__ gru, const float* __restrict__ query,
    const __bf16* __restrict__ wsA1, const float* __restrict__ aw2,
    const float* __restrict__ ab1, const float* __restrict__ ab2,
    const float* __restrict__ aw3, const float* __restrict__ ab3,
    float* __restrict__ logits) {
  __shared__ __align__(16) __bf16 sA1[16384];  // 32 KB swizzled B operands
  __shared__ float sAw2[512];
  __shared__ float sAb1[32], sAb2[16], sAw3[16];
  __shared__ float sAb3;
  __shared__ float sH1[8][16 * 33];            // per-wave h1 transpose scratch

  // async stage the swizzled aw1 B-operands into LDS
  for (int i = threadIdx.x; i < 2048; i += 256)
    async_copy16((char*)sA1 + i * 16, (const char*)wsA1 + i * 16);
  for (int i = threadIdx.x; i < 512; i += 256) sAw2[i] = aw2[i];
  if (threadIdx.x < 32) sAb1[threadIdx.x] = ab1[threadIdx.x];
  if (threadIdx.x < 16) {
    sAb2[threadIdx.x] = ab2[threadIdx.x];
    sAw3[threadIdx.x] = aw3[threadIdx.x];
  }
  if (threadIdx.x == 0) sAb3 = ab3[0];
  wait_async0();
  __syncthreads();

  const int wid = threadIdx.x >> 5, lane = threadIdx.x & 31;
  const int Lhi = lane >> 4, ln = lane & 15;
  const int tile = blockIdx.x * 8 + wid;                // < 51200
  const long rid = (long)tile * 16 + ln;                // flattened (b,t) row
  const float* qrow = query + (rid / Tt) * Ee;
  const float* krow = gru + rid * Ee;

  // Per-lane K-runs of the A layout: 4 blocks of 8 contiguous features.
  float q8[4][8], k8[4][8];
#pragma unroll
  for (int blk = 0; blk < 4; ++blk) {
    int off = blk * 16 + Lhi * 8;
    float4 qa = *(const float4*)(qrow + off);
    float4 qb = *(const float4*)(qrow + off + 4);
    float4 ca = *(const float4*)(krow + off);
    float4 cb = *(const float4*)(krow + off + 4);
    q8[blk][0] = qa.x; q8[blk][1] = qa.y; q8[blk][2] = qa.z; q8[blk][3] = qa.w;
    q8[blk][4] = qb.x; q8[blk][5] = qb.y; q8[blk][6] = qb.z; q8[blk][7] = qb.w;
    k8[blk][0] = ca.x; k8[blk][1] = ca.y; k8[blk][2] = ca.z; k8[blk][3] = ca.w;
    k8[blk][4] = cb.x; k8[blk][5] = cb.y; k8[blk][6] = cb.z; k8[blk][7] = cb.w;
  }

  v8f acc0 = {}, acc1 = {};
  const v16bf* Ba = (const v16bf*)sA1;
#pragma unroll
  for (int c = 0; c < 8; ++c) {
    const int s = c >> 1;                 // feature section: q, k, q-k, q*k
    const int i0 = 2 * (c & 1), i1 = i0 + 1;
    v16bf fa;
#pragma unroll
    for (int i = 0; i < 8; ++i) {
      float qv = q8[i0][i], kv = k8[i0][i];
      float v = (s == 0) ? qv : (s == 1) ? kv : (s == 2) ? (qv - kv) : (qv * kv);
      fa[i] = (__bf16)v;
      float qv2 = q8[i1][i], kv2 = k8[i1][i];
      float v2 = (s == 0) ? qv2 : (s == 1) ? kv2 : (s == 2) ? (qv2 - kv2) : (qv2 * kv2);
      fa[i + 8] = (__bf16)v2;
    }
    v16bf b0 = Ba[(0 * 8 + c) * 32 + lane];
    v16bf b1 = Ba[(1 * 8 + c) * 32 + lane];
    acc0 = WMMA_BF16(fa, b0, acc0);
    acc1 = WMMA_BF16(fa, b1, acc1);
  }

  // sigmoid + transpose via LDS (D layout: lane=col, vgpr r -> row r+8*Lhi)
  float* h1 = sH1[wid];
#pragma unroll
  for (int r = 0; r < 8; ++r) {
    int m = r + 8 * Lhi;
    h1[m * 33 + ln]      = sigf(acc0[r] + sAb1[ln]);
    h1[m * 33 + 16 + ln] = sigf(acc1[r] + sAb1[16 + ln]);
  }
  __syncthreads();

  if (lane < 16) {
    const int m = lane;
    float lg = sAb3;
#pragma unroll 4
    for (int nn = 0; nn < 16; ++nn) {
      float a = sAb2[nn];
      for (int kk = 0; kk < 32; ++kk) a += h1[m * 33 + kk] * sAw2[kk * 16 + nn];
      lg += sigf(a) * sAw3[nn];
    }
    logits[tile * 16 + m] = lg;
  }
}

// ---------------------------------------------------------------------------
// Kernel 2: masked softmax over T, in place (logits -> attention weights).
// One wave32 per batch row.
// ---------------------------------------------------------------------------
__global__ __launch_bounds__(256) void softmax_mask(
    float* __restrict__ logits, const int* __restrict__ mask) {
  const int wid = threadIdx.x >> 5, lane = threadIdx.x & 31;
  const int b = blockIdx.x * 8 + wid;  // < 4096
  const long base = (long)b * Tt;
  float lv[7];
  float mx = -3.4e38f;
#pragma unroll
  for (int i = 0; i < 7; ++i) {
    int t = lane + i * 32;
    float l = -3.4e38f;
    if (t < Tt) {
      l = logits[base + t];
      if (mask[base + t] == 0) l = -1.0e9f;
    }
    lv[i] = l;
    mx = fmaxf(mx, l);
  }
#pragma unroll
  for (int off = 16; off; off >>= 1) mx = fmaxf(mx, __shfl_xor(mx, off, 32));
  float sum = 0.0f;
#pragma unroll
  for (int i = 0; i < 7; ++i) {
    int t = lane + i * 32;
    float e = (t < Tt) ? __expf(lv[i] - mx) : 0.0f;
    lv[i] = e;
    sum += e;
  }
#pragma unroll
  for (int off = 16; off; off >>= 1) sum += __shfl_xor(sum, off, 32);
  float inv = 1.0f / sum;
#pragma unroll
  for (int i = 0; i < 7; ++i) {
    int t = lane + i * 32;
    if (t < Tt) logits[base + t] = lv[i] * inv;
  }
}

// ---------------------------------------------------------------------------
// Kernel 3: fused GRU scan. One wave per 16 batch rows; h in padded LDS.
// Per step: 48x v_wmma_f32_16x16x32_bf16 (k@W and h@U, N=192 in 12 tiles).
// Keys are register double-buffered one step ahead; att/mask staged into LDS
// 8 timesteps at a time via double-buffered global_load_async_to_lds_b128.
// ---------------------------------------------------------------------------
struct K8 { float4 a[8]; };

__device__ __forceinline__ K8 load_k8(const float* row, int Lhi) {
  K8 r;
#pragma unroll
  for (int c = 0; c < 2; ++c) {
    const int cb = c * 32 + 8 * Lhi;
    r.a[c * 4 + 0] = *(const float4*)(row + cb);
    r.a[c * 4 + 1] = *(const float4*)(row + cb + 4);
    r.a[c * 4 + 2] = *(const float4*)(row + cb + 16);
    r.a[c * 4 + 3] = *(const float4*)(row + cb + 20);
  }
  return r;
}

__device__ __forceinline__ void cvt_k8(const K8& r, v16bf* out) {
#pragma unroll
  for (int c = 0; c < 2; ++c) {
    float4 x0 = r.a[c * 4 + 0], x1 = r.a[c * 4 + 1];
    float4 x2 = r.a[c * 4 + 2], x3 = r.a[c * 4 + 3];
    v16bf v;
    v[0] = (__bf16)x0.x; v[1] = (__bf16)x0.y; v[2] = (__bf16)x0.z; v[3] = (__bf16)x0.w;
    v[4] = (__bf16)x1.x; v[5] = (__bf16)x1.y; v[6] = (__bf16)x1.z; v[7] = (__bf16)x1.w;
    v[8] = (__bf16)x2.x; v[9] = (__bf16)x2.y; v[10] = (__bf16)x2.z; v[11] = (__bf16)x2.w;
    v[12] = (__bf16)x3.x; v[13] = (__bf16)x3.y; v[14] = (__bf16)x3.z; v[15] = (__bf16)x3.w;
    out[c] = v;
  }
}

// Issue async loads of 8 timesteps of att (fp32) + mask (int) for 16 rows.
__device__ __forceinline__ void stage_chunk(const float* att, const int* mask,
                                            int b0, int tbase, float* attbuf,
                                            int* maskbuf, int lane) {
  const int m = lane >> 1, q = (lane & 1) * 4;
  const long off = (long)(b0 + m) * Tt + tbase + q;
  async_copy16((char*)attbuf + lane * 16, att + off);
  async_copy16((char*)maskbuf + lane * 16, mask + off);
}

__global__ __launch_bounds__(64) void gru_scan(
    const float* __restrict__ gru, const float* __restrict__ att,
    const int* __restrict__ mask, const __bf16* __restrict__ wsW,
    const __bf16* __restrict__ wsU, const float* __restrict__ bias,
    float* __restrict__ out) {
  __shared__ __align__(16) __bf16 sW[12288];   // 24 KB swizzled B operands
  __shared__ __align__(16) __bf16 sU[12288];   // 24 KB
  __shared__ float sB[192];
  __shared__ float sH[2][16 * 68];             // padded h state, per wave
  __shared__ __align__(16) float sAtt[2][2][128];  // [wave][buf][m*8+ti]
  __shared__ __align__(16) int   sMsk[2][2][128];

  // async weight staging
  for (int i = threadIdx.x; i < 1536; i += 64)
    async_copy16((char*)sW + i * 16, (const char*)wsW + i * 16);
  for (int i = threadIdx.x; i < 1536; i += 64)
    async_copy16((char*)sU + i * 16, (const char*)wsU + i * 16);
  for (int i = threadIdx.x; i < 192; i += 64) sB[i] = bias[i];

  const int wid = threadIdx.x >> 5, lane = threadIdx.x & 31;
  const int Lhi = lane >> 4, ln = lane & 15;
  float* h = sH[wid];
  for (int i = lane; i < 16 * 68; i += 32) h[i] = 0.0f;
  wait_async0();
  __syncthreads();

  const int b0 = (blockIdx.x * 2 + wid) * 16;
  const float* krow = gru + (long)(b0 + ln) * Tt * Ee;  // advances by Ee per t
  const v16bf* BW = (const v16bf*)sW;
  const v16bf* BU = (const v16bf*)sU;

  // prologue: stage att/mask chunk 0, load keys for t=0
  stage_chunk(att, mask, b0, 0, sAtt[wid][0], sMsk[wid][0], lane);
  K8 kcur = load_k8(krow, Lhi);

  for (int tc = 0; tc < 25; ++tc) {            // 25 chunks x 8 timesteps
    const int buf = tc & 1;
    if (tc < 24) {
      stage_chunk(att, mask, b0, (tc + 1) * 8, sAtt[wid][buf ^ 1],
                  sMsk[wid][buf ^ 1], lane);
      wait_async2();                           // chunk tc complete (in-order)
    } else {
      wait_async0();
    }
    const float* attb = sAtt[wid][buf];
    const int* mskb = sMsk[wid][buf];

#pragma unroll 1
    for (int ti = 0; ti < 8; ++ti) {
      const int t = tc * 8 + ti;
      __builtin_prefetch(krow + 2 * Ee, 0, 0);      // global_prefetch 2 ahead
      // software-pipelined load of next timestep's keys
      const float* nrow = (t + 1 < Tt) ? (krow + Ee) : krow;
      K8 knext = load_k8(nrow, Lhi);

      v16bf ka[2];
      cvt_k8(kcur, ka);

      // h A operands from LDS (A layout: lane=row, slots=K runs)
      v16bf ha[2];
#pragma unroll
      for (int c = 0; c < 2; ++c) {
        const int cb = c * 32 + 8 * Lhi;
        const float* hlrow = h + ln * 68;
        float4 b0v = *(const float4*)(hlrow + cb);
        float4 b1v = *(const float4*)(hlrow + cb + 4);
        float4 b2v = *(const float4*)(hlrow + cb + 16);
        float4 b3v = *(const float4*)(hlrow + cb + 20);
        v16bf w;
        w[0] = (__bf16)b0v.x; w[1] = (__bf16)b0v.y; w[2] = (__bf16)b0v.z; w[3] = (__bf16)b0v.w;
        w[4] = (__bf16)b1v.x; w[5] = (__bf16)b1v.y; w[6] = (__bf16)b1v.z; w[7] = (__bf16)b1v.w;
        w[8] = (__bf16)b2v.x; w[9] = (__bf16)b2v.y; w[10] = (__bf16)b2v.z; w[11] = (__bf16)b2v.w;
        w[12] = (__bf16)b3v.x; w[13] = (__bf16)b3v.y; w[14] = (__bf16)b3v.z; w[15] = (__bf16)b3v.w;
        ha[c] = w;
      }
      // attention weights + mask for the 8 rows this lane owns in D layout
      float av[8];
      int mv[8];
#pragma unroll
      for (int r = 0; r < 8; ++r) {
        int m = r + 8 * Lhi;
        av[r] = attb[m * 8 + ti];
        mv[r] = mskb[m * 8 + ti];
      }
#pragma unroll
      for (int j = 0; j < 4; ++j) {
        const int ntr = j, ntz = 4 + j, ntn = 8 + j;
        v8f axr = {}, axz = {}, axn = {}, ahr = {}, ahz = {}, ahn = {};
        axr = WMMA_BF16(ka[0], BW[(ntr * 2 + 0) * 32 + lane], axr);
        axr = WMMA_BF16(ka[1], BW[(ntr * 2 + 1) * 32 + lane], axr);
        axz = WMMA_BF16(ka[0], BW[(ntz * 2 + 0) * 32 + lane], axz);
        axz = WMMA_BF16(ka[1], BW[(ntz * 2 + 1) * 32 + lane], axz);
        axn = WMMA_BF16(ka[0], BW[(ntn * 2 + 0) * 32 + lane], axn);
        axn = WMMA_BF16(ka[1], BW[(ntn * 2 + 1) * 32 + lane], axn);
        ahr = WMMA_BF16(ha[0], BU[(ntr * 2 + 0) * 32 + lane], ahr);
        ahr = WMMA_BF16(ha[1], BU[(ntr * 2 + 1) * 32 + lane], ahr);
        ahz = WMMA_BF16(ha[0], BU[(ntz * 2 + 0) * 32 + lane], ahz);
        ahz = WMMA_BF16(ha[1], BU[(ntz * 2 + 1) * 32 + lane], ahz);
        ahn = WMMA_BF16(ha[0], BU[(ntn * 2 + 0) * 32 + lane], ahn);
        ahn = WMMA_BF16(ha[1], BU[(ntn * 2 + 1) * 32 + lane], ahn);

        const int col = j * 16 + ln;
        const float br = sB[col], bz = sB[64 + col], bn = sB[128 + col];
#pragma unroll
        for (int r = 0; r < 8; ++r) {
          int m = r + 8 * Lhi;
          float rg = sigf(axr[r] + br + ahr[r]);
          float zg = sigf(axz[r] + bz + ahz[r]);
          float ng = tanhfast(axn[r] + bn + rg * ahn[r]);
          float zt = av[r] * zg;
          float hold = h[m * 68 + col];
          float hnew = fmaf(zt, ng - hold, hold);   // (1-zt)*h + zt*n
          h[m * 68 + col] = mv[r] ? hnew : hold;
        }
      }
      kcur = knext;
      krow += Ee;
    }
  }
  // write h_final
#pragma unroll
  for (int r = 0; r < 8; ++r) {
    int m = r + 8 * Lhi;
#pragma unroll
    for (int j = 0; j < 4; ++j)
      out[(long)(b0 + m) * Ee + j * 16 + ln] = h[m * 68 + j * 16 + ln];
  }
}

// ---------------------------------------------------------------------------
extern "C" void kernel_launch(void* const* d_in, const int* in_sizes, int n_in,
                              void* d_out, int out_size, void* d_ws, size_t ws_size,
                              hipStream_t stream) {
  (void)in_sizes; (void)n_in; (void)out_size; (void)ws_size;
  const float* gru   = (const float*)d_in[0];   // (B,T,E)
  const float* query = (const float*)d_in[1];   // (B,E)
  const int*   mask  = (const int*)d_in[2];     // (B,T)
  const float* aw1   = (const float*)d_in[3];   // (256,32)
  const float* ab1   = (const float*)d_in[4];
  const float* aw2   = (const float*)d_in[5];   // (32,16)
  const float* ab2   = (const float*)d_in[6];
  const float* aw3   = (const float*)d_in[7];   // (16,1)
  const float* ab3   = (const float*)d_in[8];
  const float* W     = (const float*)d_in[9];   // (64,192)
  const float* U     = (const float*)d_in[10];  // (64,192)
  const float* bias  = (const float*)d_in[11];  // (192,)
  float* out = (float*)d_out;                   // (B,E)

  // workspace layout
  float*  logits = (float*)d_ws;                               // B*T floats
  __bf16* wsA1   = (__bf16*)((char*)d_ws + (size_t)Bb * Tt * 4);
  __bf16* wsW    = wsA1 + 16384;
  __bf16* wsU    = wsW + 12288;

  prep_swizzle<<<160, 256, 0, stream>>>(aw1, W, U, wsA1, wsW, wsU);
  attn_logits<<<6400, 256, 0, stream>>>(gru, query, wsA1, aw2, ab1, ab2, aw3,
                                        ab3, logits);
  softmax_mask<<<512, 256, 0, stream>>>(logits, mask);
  gru_scan<<<128, 64, 0, stream>>>(gru, logits, mask, wsW, wsU, bias, out);
}
